// GMMConvNet_62414464745577
// MI455X (gfx1250) — matrix operand order repaired and checked by reading, not compiled, not run on hardware
//
#include <hip/hip_runtime.h>

#define NNODES 10000
#define NEDGES 60000
#define KK 25            // Gaussian kernels
#define NLAYERS 12

typedef __attribute__((ext_vector_type(16))) __bf16 v16bf;
typedef __attribute__((ext_vector_type(8)))  float  v8f;

union F16x16 { v16bf v; unsigned short u[16]; };
union F32x8  { v8f v; float f[8]; };

__device__ __forceinline__ unsigned short f2bf(float x) {
  unsigned int u = __float_as_uint(x);
  if ((u & 0x7fffffffu) > 0x7f800000u) return (unsigned short)((u >> 16) | 0x40u); // NaN
  unsigned int r = u + 0x7fffu + ((u >> 16) & 1u);                                 // RNE
  return (unsigned short)(r >> 16);
}

// ---------------- CSR construction ----------------
__global__ void k_zero_i32(int* p, int n) {
  int i = blockIdx.x * blockDim.x + threadIdx.x;
  if (i < n) p[i] = 0;
}

__global__ void k_deg(const int* __restrict__ ei, int E, int* __restrict__ deg) {
  int e = blockIdx.x * blockDim.x + threadIdx.x;
  if (e < E) atomicAdd(&deg[ei[E + e]], 1);   // row 1 of edge_index = dst
}

// Single-block exclusive scan over N entries; also writes cursor copy + inv_deg.
__global__ void k_scan(const int* __restrict__ deg, int* __restrict__ row_ptr,
                       int* __restrict__ cursor, float* __restrict__ inv_deg, int n) {
  __shared__ int buf[1024];
  __shared__ int carry;
  int tid = threadIdx.x;
  if (tid == 0) carry = 0;
  __syncthreads();
  for (int base = 0; base < n; base += 1024) {
    int i = base + tid;
    int v = (i < n) ? deg[i] : 0;
    buf[tid] = v;
    __syncthreads();
    #pragma unroll
    for (int off = 1; off < 1024; off <<= 1) {
      int t = (tid >= off) ? buf[tid - off] : 0;
      __syncthreads();
      buf[tid] += t;
      __syncthreads();
    }
    int incl = buf[tid];
    if (i < n) {
      int rp = carry + incl - v;            // exclusive
      row_ptr[i] = rp;
      cursor[i]  = rp;
      inv_deg[i] = 1.0f / fmaxf((float)v, 1.0f);
    }
    __syncthreads();
    if (tid == 1023) carry += incl;
    __syncthreads();
  }
  if (tid == 0) row_ptr[n] = carry;
}

__global__ void k_fill(const int* __restrict__ ei, int* __restrict__ cursor,
                       int* __restrict__ slot, int E) {
  int e = blockIdx.x * blockDim.x + threadIdx.x;
  if (e >= E) return;
  int d = ei[E + e];
  int pos = atomicAdd(&cursor[d], 1);
  slot[pos] = e;
}

// ---------------- per-layer kernels ----------------
__global__ void k_pad_x(const float* __restrict__ x, float* __restrict__ xp,
                        int n, int Cin, int Cinp) {
  int idx = blockIdx.x * blockDim.x + threadIdx.x;
  if (idx >= n * Cinp) return;
  int c = idx % Cinp, r = idx / Cinp;
  xp[idx] = (c < Cin) ? x[(long)r * Cin + c] : 0.0f;
}

// w[e,k] = exp(sum_d -0.5*(u-mu)^2 / (eps + sigma^2))
__global__ void k_edgew(const float* __restrict__ ea, const float* __restrict__ mu,
                        const float* __restrict__ sigma, float* __restrict__ w, int E) {
  int idx = blockIdx.x * blockDim.x + threadIdx.x;
  if (idx >= E * KK) return;
  int e = idx / KK, k = idx - e * KK;
  float s = 0.0f;
  #pragma unroll
  for (int d = 0; d < 3; ++d) {
    float diff = ea[e * 3 + d] - mu[k * 3 + d];
    float sg = sigma[k * 3 + d];
    s += -0.5f * diff * diff / (1e-15f + sg * sg);
  }
  w[idx] = expf(s);
}

// One block per node. Thread i accumulates channel i across incoming edges for all
// 25 kernels (registers), then writes the bf16 A row:
//   A[n, k*Cinp + i] = inv_deg[n] * agg   (k < 25)
//   A[n, 25*Cinp + i] = x[n, i]           (root-term block)
__global__ __launch_bounds__(256)
void k_agg(const float* __restrict__ xp, const float* __restrict__ w,
           const int* __restrict__ srcrow, const int* __restrict__ row_ptr,
           const int* __restrict__ slot, const float* __restrict__ inv_deg,
           unsigned short* __restrict__ A, int Cinp) {
  int n = blockIdx.x;
  int i = threadIdx.x;
  __shared__ float wk[KK];
  __shared__ int s_src;
  float acc[KK];
  #pragma unroll
  for (int k = 0; k < KK; ++k) acc[k] = 0.0f;

  int e0 = row_ptr[n], e1 = row_ptr[n + 1];
  for (int idx = e0; idx < e1; ++idx) {
    __syncthreads();
    if (i < KK) {
      int e = slot[idx];
      wk[i] = w[(long)e * KK + i];
      if (i == 0) s_src = srcrow[e];
    }
    __syncthreads();
    if (i < Cinp) {
      float xs = xp[(long)s_src * Cinp + i];
      #pragma unroll
      for (int k = 0; k < KK; ++k) acc[k] += wk[k] * xs;
    }
  }
  if (i < Cinp) {
    float sc = inv_deg[n];
    long base = (long)n * (KK + 1) * Cinp;
    #pragma unroll
    for (int k = 0; k < KK; ++k) A[base + (long)k * Cinp + i] = f2bf(acc[k] * sc);
    A[base + (long)KK * Cinp + i] = f2bf(xp[(long)n * Cinp + i]);
  }
}

// Pack [G (K,Cin,Co); root (Cin,Co)] -> bf16 Bt in a K-tiled, transpose-free layout:
//   Bt[kt][col][k]  with kt = K-tile index (32 K per tile), element (kt*32+k, col).
// This lets the GEMM stage B with async b128 copies directly (no LDS transpose).
__global__ void k_weights(const float* __restrict__ G, const float* __restrict__ root,
                          unsigned short* __restrict__ Bt, int Cin, int Co,
                          int Cinp, int Cop) {
  long idx = (long)blockIdx.x * blockDim.x + threadIdx.x;
  long Kdim = (long)(KK + 1) * Cinp;
  long total = Kdim * Cop;
  if (idx >= total) return;
  int k  = (int)(idx & 31);
  long t = idx >> 5;
  int o  = (int)(t % Cop);            // output column
  int kt = (int)(t / Cop);            // K-tile
  long kkg = (long)kt * 32 + k;       // global K index
  int i  = (int)(kkg % Cinp);
  int kb = (int)(kkg / Cinp);
  float v = 0.0f;
  if (i < Cin && o < Co)
    v = (kb < KK) ? G[((long)kb * Cin + i) * Co + o] : root[(long)i * Co + o];
  Bt[idx] = f2bf(v);
}

// WMMA bf16 GEMM: out[N,Cop] = A[N,Kdim] x Bt, fused bias(+ELU+BN).
// Block tile 128x64 (8 waves, 32x32 per wave => 4 v_wmma per K-step).
// BOTH operands staged with CDNA5 async global->LDS copies (ASYNCcnt),
// double-buffered LDS, one s_wait_asynccnt + barrier per K-step.
#define BM 128
#define BN 64
#define BK 32
#define LDP (BK + 16)   // LDS row stride in shorts; keeps 16B alignment for b128

__global__ __launch_bounds__(256)
void k_gemm(const unsigned short* __restrict__ A, const unsigned short* __restrict__ Bt,
            float* __restrict__ outp, const float* __restrict__ bias,
            const float* __restrict__ gamma, const float* __restrict__ beta,
            const float* __restrict__ rm, const float* __restrict__ rv,
            int Nrow, int Kdim, int Co, int Cop, int fuse_bn) {
  __shared__ unsigned short As[2][BM][LDP];
  __shared__ unsigned short Bs[2][BN][LDP];

  int tid  = threadIdx.x;
  int lane = tid & 31;
  int wave = tid >> 5;
  int bm = blockIdx.x * BM;
  int bn = blockIdx.y * BN;
  int wm = (wave >> 1) * 32;   // wave row offset (4 row-groups)
  int wn = (wave & 1) * 32;    // wave col offset (2 col-groups)
  int mlo = lane & 15;
  int hi  = lane >> 4;

  // Async staging indices. All lanes always issue (EXEC-masked lanes would leave
  // stale LDS), so out-of-range rows/cols are clamped; clamped data only feeds
  // output rows/cols the epilogue never writes.
  int ar[2], ac[2]; long agr[2];
  unsigned alds[2][2];                    // A: [buf][chunk] LDS byte offsets
  #pragma unroll
  for (int j = 0; j < 2; ++j) {
    int ch = tid + 256 * j;
    ar[j] = ch >> 2;
    ac[j] = (ch & 3) * 8;
    int gr = bm + ar[j];
    agr[j] = (gr < Nrow) ? gr : (Nrow - 1);
    alds[0][j] = (unsigned)(size_t)&As[0][ar[j]][ac[j]];
    alds[1][j] = (unsigned)(size_t)&As[1][ar[j]][ac[j]];
  }
  int bn_l = tid >> 2;                    // B: LDS row 0..63
  int bko  = (tid & 3) * 8;               // B: k offset within tile
  long bcol = bn + bn_l;
  if (bcol >= Cop) bcol = Cop - 1;
  unsigned blds[2] = { (unsigned)(size_t)&Bs[0][bn_l][bko],
                       (unsigned)(size_t)&Bs[1][bn_l][bko] };

  // Issue async loads for tile 0 -> buffer 0 (A: 2 b128/thread, B: 1 b128/thread).
  #pragma unroll
  for (int j = 0; j < 2; ++j) {
    const unsigned short* gp = A + agr[j] * Kdim + ac[j];
    asm volatile("global_load_async_to_lds_b128 %0, %1, off"
                 :: "v"(alds[0][j]), "v"(gp) : "memory");
  }
  {
    const unsigned short* gp = Bt + bcol * 32 + bko;   // tile 0
    asm volatile("global_load_async_to_lds_b128 %0, %1, off"
                 :: "v"(blds[0]), "v"(gp) : "memory");
  }

  F32x8 c[2][2];
  #pragma unroll
  for (int a = 0; a < 2; ++a)
    #pragma unroll
    for (int b = 0; b < 2; ++b)
      #pragma unroll
      for (int r = 0; r < 8; ++r) c[a][b].f[r] = 0.0f;

  int nIter = Kdim / BK;                  // Kdim is always a multiple of 32
  int cur = 0;
  for (int it = 0; it < nIter; ++it) {
    // Wait for this wave's async copies of tile(it), then make LDS visible.
    asm volatile("s_wait_asynccnt 0x0" ::: "memory");
    __syncthreads();
    // Prefetch tile(it+1) into the alternate buffers (overlaps with WMMA below;
    // those buffers were last read at iteration it-1, before this barrier).
    if (it + 1 < nIter) {
      int ktn = (it + 1) * BK;
      #pragma unroll
      for (int j = 0; j < 2; ++j) {
        const unsigned short* gp = A + agr[j] * Kdim + ktn + ac[j];
        asm volatile("global_load_async_to_lds_b128 %0, %1, off"
                     :: "v"(alds[cur ^ 1][j]), "v"(gp) : "memory");
      }
      const unsigned short* gp = Bt + ((long)(it + 1) * Cop + bcol) * 32 + bko;
      asm volatile("global_load_async_to_lds_b128 %0, %1, off"
                   :: "v"(blds[cur ^ 1]), "v"(gp) : "memory");
    }

    // Build fragments per ISA 7.12.2 wave32 layouts and issue 4 WMMAs.
    F16x16 af[2], bfr[2];
    #pragma unroll
    for (int t = 0; t < 2; ++t) {
      int m = wm + t * 16 + mlo;
      #pragma unroll
      for (int e = 0; e < 16; ++e) {
        int k = hi * 8 + (e >> 3) * 16 + (e & 7);   // A: {0-7,16-23} / {8-15,24-31}
        af[t].u[e] = As[cur][m][k];
      }
      int nn = wn + t * 16 + mlo;
      #pragma unroll
      for (int e = 0; e < 16; ++e) {
        int k = hi * 8 + (e >> 3) * 16 + (e & 7);   // B: {0-7,16-23} / {8-15,24-31}
        bfr[t].u[e] = Bs[cur][nn][k];
      }
    }
    #pragma unroll
    for (int a = 0; a < 2; ++a)
      #pragma unroll
      for (int b = 0; b < 2; ++b)
        c[a][b].v = __builtin_amdgcn_wmma_f32_16x16x32_bf16(
            false, af[a].v, false, bfr[b].v, (short)0, c[a][b].v, false, false);
    cur ^= 1;
  }

  // Epilogue: bias + (ELU + BatchNorm eval) fused; zero channel padding.
  #pragma unroll
  for (int a = 0; a < 2; ++a) {
    #pragma unroll
    for (int b = 0; b < 2; ++b) {
      #pragma unroll
      for (int r = 0; r < 8; ++r) {
        int row = bm + wm + a * 16 + hi * 8 + r;     // C/D: VGPR r -> M=r or 8+r
        int col = bn + wn + b * 16 + mlo;
        if (row < Nrow && col < Cop) {
          float v = 0.0f;
          if (col < Co) {
            v = c[a][b].f[r] + bias[col];
            if (fuse_bn) {
              v = v > 0.0f ? v : (expf(v) - 1.0f);                     // ELU
              v = (v - rm[col]) * gamma[col] * rsqrtf(rv[col] + 1e-5f) + beta[col];
            }
          }
          outp[(long)row * Cop + col] = v;
        }
      }
    }
  }
}

__global__ void k_copy_out(const float* __restrict__ xp, float* __restrict__ out,
                           int n, int Co, int Cinp) {
  int idx = blockIdx.x * blockDim.x + threadIdx.x;
  if (idx >= n * Co) return;
  int r = idx / Co, cc = idx - r * Co;
  out[idx] = xp[(long)r * Cinp + cc];
}

// ---------------- host orchestration ----------------
extern "C" void kernel_launch(void* const* d_in, const int* in_sizes, int n_in,
                              void* d_out, int out_size, void* d_ws, size_t ws_size,
                              hipStream_t stream) {
  static const int CH[NLAYERS + 1]  = {50, 75, 100, 200, 200, 200, 200, 200, 200, 200, 100, 75, 50};
  static const int CHP[NLAYERS + 1] = {64, 80, 112, 208, 208, 208, 208, 208, 208, 208, 112, 80, 64};
  const int N = NNODES, E = NEDGES;

  const float* x  = (const float*)d_in[0];
  const float* ea = (const float*)d_in[1];
  const int*   ei = (const int*)d_in[n_in - 1];   // edge_index is the last input

  struct P { const float *G, *mu, *sigma, *root, *bias, *gamma, *beta, *rm, *rv; };
  P p[NLAYERS];
  int b = 2;   // params flattened in dict insertion order per layer
  for (int i = 0; i < NLAYERS; ++i) {
    p[i].G     = (const float*)d_in[b + 0];
    p[i].mu    = (const float*)d_in[b + 1];
    p[i].sigma = (const float*)d_in[b + 2];
    p[i].root  = (const float*)d_in[b + 3];
    p[i].bias  = (const float*)d_in[b + 4];
    if (i < NLAYERS - 1) {
      p[i].gamma = (const float*)d_in[b + 5];
      p[i].beta  = (const float*)d_in[b + 6];
      p[i].rm    = (const float*)d_in[b + 7];
      p[i].rv    = (const float*)d_in[b + 8];
      b += 9;
    } else {
      p[i].gamma = p[i].beta = p[i].rm = p[i].rv = p[i].bias;  // unused (fuse_bn=0)
      b += 5;
    }
  }

  // Workspace carve-up (256B aligned).
  char* wsp = (char*)d_ws;
  auto alloc = [&](size_t bytes) -> void* {
    void* r = (void*)wsp;
    wsp += (bytes + 255) & ~(size_t)255;
    return r;
  };
  int*   deg     = (int*)alloc((size_t)N * 4);
  int*   row_ptr = (int*)alloc((size_t)(N + 1) * 4);
  int*   cursor  = (int*)alloc((size_t)N * 4);
  int*   slot    = (int*)alloc((size_t)E * 4);
  float* inv_deg = (float*)alloc((size_t)N * 4);
  float* wbuf    = (float*)alloc((size_t)E * KK * 4);
  float* xp0     = (float*)alloc((size_t)N * 208 * 4);
  float* xp1     = (float*)alloc((size_t)N * 208 * 4);
  unsigned short* Abuf = (unsigned short*)alloc((size_t)N * (KK + 1) * 208 * 2);
  unsigned short* Bbuf = (unsigned short*)alloc((size_t)(KK + 1) * 208 * 208 * 2);
  (void)ws_size; (void)in_sizes; (void)out_size;

  // CSR + inv_deg (once; shared across layers).
  k_zero_i32<<<(N + 255) / 256, 256, 0, stream>>>(deg, N);
  k_deg<<<(E + 255) / 256, 256, 0, stream>>>(ei, E, deg);
  k_scan<<<1, 1024, 0, stream>>>(deg, row_ptr, cursor, inv_deg, N);
  k_fill<<<(E + 255) / 256, 256, 0, stream>>>(ei, cursor, slot, E);

  // x -> padded f32 [N, 64]
  k_pad_x<<<(N * CHP[0] + 255) / 256, 256, 0, stream>>>(x, xp0, N, CH[0], CHP[0]);

  float* xa = xp0;
  float* xb = xp1;
  for (int i = 0; i < NLAYERS; ++i) {
    int Cin = CH[i], Co = CH[i + 1], Cinp = CHP[i], Cop = CHP[i + 1];
    int Kdim = (KK + 1) * Cinp;

    k_edgew<<<(E * KK + 255) / 256, 256, 0, stream>>>(ea, p[i].mu, p[i].sigma, wbuf, E);
    k_agg<<<N, 256, 0, stream>>>(xa, wbuf, ei, row_ptr, slot, inv_deg, Abuf, Cinp);
    {
      long tot = (long)Kdim * Cop;
      k_weights<<<(int)((tot + 255) / 256), 256, 0, stream>>>(p[i].G, p[i].root, Bbuf,
                                                              Cin, Co, Cinp, Cop);
    }
    dim3 grid((N + BM - 1) / BM, (Cop + BN - 1) / BN);
    k_gemm<<<grid, 256, 0, stream>>>(Abuf, Bbuf, xb, p[i].bias, p[i].gamma, p[i].beta,
                                     p[i].rm, p[i].rv, N, Kdim, Co, Cop,
                                     (i < NLAYERS - 1) ? 1 : 0);
    float* t = xa; xa = xb; xb = t;
  }

  k_copy_out<<<(N * CH[NLAYERS] + 255) / 256, 256, 0, stream>>>(xa, (float*)d_out, N,
                                                                CH[NLAYERS], CHP[NLAYERS]);
}
